// SparseTransformerLayer_42674795053797
// MI455X (gfx1250) — matrix operand order repaired and checked by reading, not compiled, not run on hardware
//
#include <hip/hip_runtime.h>
#include <hip/hip_bf16.h>

#define DEV __device__ __forceinline__

typedef __attribute__((ext_vector_type(16))) _Float16 v16h;
typedef __attribute__((ext_vector_type(8)))  _Float16 v8h;
typedef __attribute__((ext_vector_type(8)))  float    v8f;

constexpr int B_  = 2;
constexpr int S_  = 1024;
constexpr int D_  = 1024;
constexpr int F_  = 4096;
constexpr int H_  = 16;
constexpr int E_  = 8;
constexpr int HD_ = D_ / H_;   // 64
constexpr int M_  = B_ * S_;   // 2048 tokens
constexpr int FH_ = F_ / 2;    // 2048
constexpr int N_QKV = 3 * D_;  // 3072

// ---------------------------------------------------------------------------
// WMMA helpers (CDNA5 wave32, v_wmma_f32_16x16x32_f16)
// ---------------------------------------------------------------------------
DEV v8f wmma_f16(v16h a, v16h b, v8f c) {
  return __builtin_amdgcn_wmma_f32_16x16x32_f16(false, a, false, b, (short)0, c,
                                                false, false);
}

// A fragment: 16x32 f16 tile at (m0,k0) from row-major matrix, leading dim ld.
DEV v16h frag_a(const _Float16* p, int ld, int m0, int k0, int lane) {
  const _Float16* q =
      p + (size_t)(m0 + (lane & 15)) * ld + k0 + ((lane >> 4) << 3);
  v8h lo = *(const v8h*)q;
  v8h hi = *(const v8h*)(q + 16);
  v16h r;
#pragma unroll
  for (int i = 0; i < 8; ++i) { r[i] = lo[i]; r[i + 8] = hi[i]; }
  return r;
}

// B fragment: 32x16 f16 tile at (k0,n0); source stores B^T row-major
// (row = n, contiguous k, leading dim ld) -> two 16B loads/lane (k 8h.., 16+8h..).
DEV v16h frag_b(const _Float16* p, int ld, int n0, int k0, int lane) {
  const _Float16* q =
      p + (size_t)(n0 + (lane & 15)) * ld + k0 + ((lane >> 4) << 3);
  v8h lo = *(const v8h*)q;
  v8h hi = *(const v8h*)(q + 16);
  v16h r;
#pragma unroll
  for (int i = 0; i < 8; ++i) { r[i] = lo[i]; r[i + 8] = hi[i]; }
  return r;
}

// Gathered A fragment for MoE: tile rows are expert-list slots.
DEV v16h frag_a_gather(const _Float16* p, int ld, const int* lst, int cnt,
                       int m0, int k0, int lane) {
  int slot = m0 + (lane & 15);
  v16h r;
  if (slot < cnt) {
    const _Float16* q =
        p + (size_t)lst[slot] * ld + k0 + ((lane >> 4) << 3);
    v8h lo = *(const v8h*)q;
    v8h hi = *(const v8h*)(q + 16);
#pragma unroll
    for (int i = 0; i < 8; ++i) { r[i] = lo[i]; r[i + 8] = hi[i]; }
  } else {
#pragma unroll
    for (int i = 0; i < 16; ++i) r[i] = (_Float16)0.0f;
  }
  return r;
}

// ---------------------------------------------------------------------------
// Small utility kernels
// ---------------------------------------------------------------------------
__global__ void k_zero_counts(int* cnt) {
  if (threadIdx.x < E_) cnt[threadIdx.x] = 0;
}

__global__ void k_cvt(const float* __restrict__ in, _Float16* __restrict__ out,
                      int n) {
  int i = blockIdx.x * blockDim.x + threadIdx.x;
  if (i < n) out[i] = (_Float16)in[i];
}

// in: (R,C) f32, out: (C,R) f16; one expert slab per blockIdx.z.
__global__ void k_cvt_T(const float* __restrict__ in, _Float16* __restrict__ out,
                        int R, int C) {
  size_t slab = (size_t)R * C;
  const float* ip = in + blockIdx.z * slab;
  _Float16* op = out + blockIdx.z * slab;
  int o = blockIdx.x * blockDim.x + threadIdx.x;  // output flat: c*R + r
  if (o < (int)slab) {
    int c = o / R, r = o - c * R;
    op[o] = (_Float16)ip[(size_t)r * C + c];
  }
}

// ---------------------------------------------------------------------------
// LayerNorm1 + RoPE -> f16 activation (one block per token row)
// ---------------------------------------------------------------------------
__global__ void k_ln1_rope(const float* __restrict__ x,
                           const float* __restrict__ w,
                           const float* __restrict__ b,
                           _Float16* __restrict__ hA) {
  __shared__ float2 red[256];
  __shared__ float nrm[D_];
  int m = blockIdx.x;
  int s = m & (S_ - 1);
  const float* row = x + (size_t)m * D_;
  float ls = 0.f, lq = 0.f;
  for (int d = threadIdx.x; d < D_; d += 256) {
    float v = row[d];
    ls += v; lq += v * v;
  }
  red[threadIdx.x] = make_float2(ls, lq);
  __syncthreads();
  for (int st = 128; st > 0; st >>= 1) {
    if ((int)threadIdx.x < st) {
      red[threadIdx.x].x += red[threadIdx.x + st].x;
      red[threadIdx.x].y += red[threadIdx.x + st].y;
    }
    __syncthreads();
  }
  float mean = red[0].x / D_;
  float var  = red[0].y / D_ - mean * mean;
  float rs   = rsqrtf(var + 1e-5f);
  for (int d = threadIdx.x; d < D_; d += 256)
    nrm[d] = (row[d] - mean) * rs * w[d] + b[d];
  __syncthreads();
  const int half = D_ / 2;
  for (int d = threadIdx.x; d < half; d += 256) {
    float freq = __expf(-(float)d * (9.210340371976184f / (float)half));
    float ang = (float)s * freq;
    float c, sn;
    __sincosf(ang, &sn, &c);
    float x1 = nrm[d], x2 = nrm[d + half];
    hA[(size_t)m * D_ + d]        = (_Float16)(x1 * c - x2 * sn);
    hA[(size_t)m * D_ + d + half] = (_Float16)(x1 * sn + x2 * c);
  }
}

// ---------------------------------------------------------------------------
// QKV GEMM: wave tile 64x32 (4x2 accumulators -> 8 WMMA / 6 frag loads per
// K-step, ~150 VGPRs: no spills). Block = 8 waves = 128x128 tile.
// Scatter epilogue into head-major Q,K and transposed V.
// ---------------------------------------------------------------------------
__global__ void __launch_bounds__(256, 1)
k_gemm_qkv(const _Float16* __restrict__ A,
           const _Float16* __restrict__ W,
           const float* __restrict__ bias,
           _Float16* __restrict__ qh,
           _Float16* __restrict__ kh,
           _Float16* __restrict__ vh) {
  int lane = threadIdx.x & 31, wv = threadIdx.x >> 5;
  int m0 = blockIdx.y * 128 + (wv & 1) * 64;
  int n0 = blockIdx.x * 128 + (wv >> 1) * 32;
  v8f acc[4][2] = {};
  for (int k0 = 0; k0 < D_; k0 += 32) {
    __builtin_prefetch(A + (size_t)(m0 + (lane & 15)) * D_ + k0 + 64, 0, 1);
    v16h af[4], bf[2];
#pragma unroll
    for (int i = 0; i < 4; ++i) af[i] = frag_a(A, D_, m0 + 16 * i, k0, lane);
#pragma unroll
    for (int j = 0; j < 2; ++j) bf[j] = frag_b(W, D_, n0 + 16 * j, k0, lane);
#pragma unroll
    for (int i = 0; i < 4; ++i)
#pragma unroll
      for (int j = 0; j < 2; ++j) acc[i][j] = wmma_f16(af[i], bf[j], acc[i][j]);
  }
  int hb = lane >> 4, ln = lane & 15;
#pragma unroll
  for (int i = 0; i < 4; ++i)
#pragma unroll
    for (int j = 0; j < 2; ++j)
#pragma unroll
      for (int r = 0; r < 8; ++r) {
        int m = m0 + i * 16 + r + 8 * hb;
        int n = n0 + j * 16 + ln;
        _Float16 val = (_Float16)(acc[i][j][r] + bias[n]);
        int third = n >> 10, wn = n & (D_ - 1);
        int head = wn >> 6, hd = wn & (HD_ - 1);
        int bb = m >> 10, sq = m & (S_ - 1);
        size_t bhs = ((size_t)(bb * H_ + head) * S_ + sq) * HD_ + hd;
        if (third == 0)      qh[bhs] = val;
        else if (third == 1) kh[bhs] = val;
        else vh[((size_t)(bb * H_ + head) * HD_ + hd) * S_ + sq] = val;
      }
}

// ---------------------------------------------------------------------------
// Attention: flash-style, one wave = 16 query rows of one (b,h).
// Q,K: [bh][s][hd]; V: [bh][hd][s] (transposed). 4 waves / block.
// ---------------------------------------------------------------------------
__global__ void __launch_bounds__(128, 1)
k_attn(const _Float16* __restrict__ qh,
       const _Float16* __restrict__ kh,
       const _Float16* __restrict__ vh,
       _Float16* __restrict__ o16) {
  __shared__ _Float16 pbuf[4][16 * 32];
  int lane = threadIdx.x & 31, wv = threadIdx.x >> 5;
  int g = blockIdx.x * 4 + wv;
  int bh = g >> 6;
  int q0 = (g & 63) << 4;
  const _Float16* Q = qh + (size_t)bh * S_ * HD_;
  const _Float16* K = kh + (size_t)bh * S_ * HD_;
  const _Float16* V = vh + (size_t)bh * HD_ * S_;
  _Float16* P = pbuf[wv];

  v16h qa0 = frag_a(Q, HD_, q0, 0, lane);
  v16h qa1 = frag_a(Q, HD_, q0, 32, lane);
  v8f o[4] = {};
  float Mx[8], L[8];
#pragma unroll
  for (int r = 0; r < 8; ++r) { Mx[r] = -1e30f; L[r] = 0.f; }
  const float scale = 0.125f;  // 1/sqrt(64)
  int hb = lane >> 4, ln = lane & 15;

  for (int c0 = 0; c0 < S_; c0 += 32) {
    v8f s0 = {}, s1 = {};
    s0 = wmma_f16(qa0, frag_b(K, HD_, c0, 0, lane), s0);
    s0 = wmma_f16(qa1, frag_b(K, HD_, c0, 32, lane), s0);
    s1 = wmma_f16(qa0, frag_b(K, HD_, c0 + 16, 0, lane), s1);
    s1 = wmma_f16(qa1, frag_b(K, HD_, c0 + 16, 32, lane), s1);
#pragma unroll
    for (int r = 0; r < 8; ++r) {
      float v0 = s0[r] * scale, v1 = s1[r] * scale;
      float mx = fmaxf(v0, v1);
#pragma unroll
      for (int d = 1; d < 16; d <<= 1) mx = fmaxf(mx, __shfl_xor(mx, d, 32));
      float Mn = fmaxf(Mx[r], mx);
      float al = __expf(Mx[r] - Mn);
      Mx[r] = Mn;
      float p0 = __expf(v0 - Mn), p1 = __expf(v1 - Mn);
      float rs = p0 + p1;
#pragma unroll
      for (int d = 1; d < 16; d <<= 1) rs += __shfl_xor(rs, d, 32);
      L[r] = L[r] * al + rs;
#pragma unroll
      for (int a = 0; a < 4; ++a) o[a][r] *= al;
      s0[r] = p0; s1[r] = p1;
    }
#pragma unroll
    for (int r = 0; r < 8; ++r) {
      P[(r + 8 * hb) * 32 + ln]      = (_Float16)s0[r];
      P[(r + 8 * hb) * 32 + 16 + ln] = (_Float16)s1[r];
    }
    __syncthreads();
    v16h pa = frag_a(P, 32, 0, 0, lane);
    __syncthreads();
#pragma unroll
    for (int a = 0; a < 4; ++a)
      o[a] = wmma_f16(pa, frag_b(V, S_, a * 16, c0, lane), o[a]);
  }
  int head = bh & (H_ - 1), bb = bh >> 4;
#pragma unroll
  for (int r = 0; r < 8; ++r) {
    float inv = 1.0f / L[r];
    int m = bb * S_ + q0 + r + 8 * hb;
#pragma unroll
    for (int a = 0; a < 4; ++a)
      o16[(size_t)m * D_ + head * HD_ + a * 16 + ln] =
          (_Float16)(o[a][r] * inv);
  }
}

// ---------------------------------------------------------------------------
// Output projection + residual: out(f32) = x + o @ w_out^T + b_out.
// Wave tile 64x32.
// ---------------------------------------------------------------------------
__global__ void __launch_bounds__(256, 1)
k_gemm_out(const _Float16* __restrict__ A,
           const _Float16* __restrict__ W,
           const float* __restrict__ bias,
           const float* __restrict__ xres,
           float* __restrict__ out) {
  int lane = threadIdx.x & 31, wv = threadIdx.x >> 5;
  int m0 = blockIdx.y * 128 + (wv & 1) * 64;
  int n0 = blockIdx.x * 128 + (wv >> 1) * 32;
  v8f acc[4][2] = {};
  for (int k0 = 0; k0 < D_; k0 += 32) {
    __builtin_prefetch(W + (size_t)(n0 + (lane & 15)) * D_ + k0 + 64, 0, 1);
    v16h af[4], bf[2];
#pragma unroll
    for (int i = 0; i < 4; ++i) af[i] = frag_a(A, D_, m0 + 16 * i, k0, lane);
#pragma unroll
    for (int j = 0; j < 2; ++j) bf[j] = frag_b(W, D_, n0 + 16 * j, k0, lane);
#pragma unroll
    for (int i = 0; i < 4; ++i)
#pragma unroll
      for (int j = 0; j < 2; ++j) acc[i][j] = wmma_f16(af[i], bf[j], acc[i][j]);
  }
  int hb = lane >> 4, ln = lane & 15;
#pragma unroll
  for (int i = 0; i < 4; ++i)
#pragma unroll
    for (int j = 0; j < 2; ++j)
#pragma unroll
      for (int r = 0; r < 8; ++r) {
        int m = m0 + i * 16 + r + 8 * hb;
        int n = n0 + j * 16 + ln;
        out[(size_t)m * D_ + n] =
            xres[(size_t)m * D_ + n] + acc[i][j][r] + bias[n];
      }
}

// ---------------------------------------------------------------------------
// LayerNorm2 + router (softmax/argmax/top-1 gate + expert lists)
// ---------------------------------------------------------------------------
__global__ void k_ln2_router(const float* __restrict__ xo,
                             const float* __restrict__ w,
                             const float* __restrict__ b,
                             const float* __restrict__ wr,
                             _Float16* __restrict__ h2,
                             float* __restrict__ logits,
                             float* __restrict__ gate,
                             int* __restrict__ cnt,
                             int* __restrict__ list) {
  __shared__ float2 red[256];
  __shared__ float nrm[D_];
  __shared__ float lg[E_];
  int m = blockIdx.x;
  const float* row = xo + (size_t)m * D_;
  float ls = 0.f, lq = 0.f;
  for (int d = threadIdx.x; d < D_; d += 256) {
    float v = row[d];
    ls += v; lq += v * v;
  }
  red[threadIdx.x] = make_float2(ls, lq);
  __syncthreads();
  for (int st = 128; st > 0; st >>= 1) {
    if ((int)threadIdx.x < st) {
      red[threadIdx.x].x += red[threadIdx.x + st].x;
      red[threadIdx.x].y += red[threadIdx.x + st].y;
    }
    __syncthreads();
  }
  float mean = red[0].x / D_;
  float var  = red[0].y / D_ - mean * mean;
  float rs   = rsqrtf(var + 1e-5f);
  for (int d = threadIdx.x; d < D_; d += 256) {
    float nv = (row[d] - mean) * rs * w[d] + b[d];
    nrm[d] = nv;
    h2[(size_t)m * D_ + d] = (_Float16)nv;
  }
  __syncthreads();
  if ((int)threadIdx.x < E_) {
    float s = 0.f;
    for (int d = 0; d < D_; ++d) s += nrm[d] * wr[d * E_ + threadIdx.x];
    lg[threadIdx.x] = s;
    logits[(size_t)m * E_ + threadIdx.x] = s;
  }
  __syncthreads();
  if (threadIdx.x == 0) {
    float mx = lg[0]; int am = 0;
    for (int e = 1; e < E_; ++e)
      if (lg[e] > mx) { mx = lg[e]; am = e; }
    float den = 0.f;
    for (int e = 0; e < E_; ++e) den += __expf(lg[e] - mx);
    gate[m] = 1.0f / den;  // softmax prob of argmax expert
    int slot = atomicAdd(&cnt[am], 1);
    list[am * M_ + slot] = m;
  }
}

// ---------------------------------------------------------------------------
// MoE FFN1: hid = h2 @ wi[e] with fused SiLU(a)*g. Wave tile 32 rows x
// (32 a-cols + paired 32 g-cols): 8 WMMA / 6 frag loads per K-step.
// ---------------------------------------------------------------------------
__global__ void __launch_bounds__(256, 1)
k_moe_ffn1(const _Float16* __restrict__ h2,
           const _Float16* __restrict__ wit,
           const int* __restrict__ cnt,
           const int* __restrict__ list,
           _Float16* __restrict__ act) {
  int e = blockIdx.z;
  int c = cnt[e];
  if ((int)blockIdx.y * 64 >= c) return;
  const int* lst = list + e * M_;
  const _Float16* Wt = wit + (size_t)e * F_ * D_;
  int lane = threadIdx.x & 31, wv = threadIdx.x >> 5;
  int m0 = blockIdx.y * 64 + (wv & 1) * 32;
  int n0 = blockIdx.x * 128 + (wv >> 1) * 32;  // 'a' column; 'g' = n0 + FH_
  v8f aa[2][2] = {}, ag[2][2] = {};
  for (int k0 = 0; k0 < D_; k0 += 32) {
    __builtin_prefetch(Wt + (size_t)(n0 + (lane & 15)) * D_ + k0 + 64, 0, 1);
    v16h af[2];
#pragma unroll
    for (int i = 0; i < 2; ++i)
      af[i] = frag_a_gather(h2, D_, lst, c, m0 + 16 * i, k0, lane);
    v16h bA[2], bG[2];
#pragma unroll
    for (int j = 0; j < 2; ++j) {
      bA[j] = frag_b(Wt, D_, n0 + 16 * j, k0, lane);
      bG[j] = frag_b(Wt, D_, n0 + FH_ + 16 * j, k0, lane);
    }
#pragma unroll
    for (int i = 0; i < 2; ++i)
#pragma unroll
      for (int j = 0; j < 2; ++j) {
        aa[i][j] = wmma_f16(af[i], bA[j], aa[i][j]);
        ag[i][j] = wmma_f16(af[i], bG[j], ag[i][j]);
      }
  }
  int hb = lane >> 4, ln = lane & 15;
#pragma unroll
  for (int i = 0; i < 2; ++i)
#pragma unroll
    for (int j = 0; j < 2; ++j)
#pragma unroll
      for (int r = 0; r < 8; ++r) {
        int slot = m0 + i * 16 + r + 8 * hb;
        if (slot < c) {
          int tok = lst[slot];
          float a = aa[i][j][r], g = ag[i][j][r];
          float s = (a / (1.0f + __expf(-a))) * g;  // SiLU(a)*g
          act[(size_t)tok * FH_ + n0 + j * 16 + ln] = (_Float16)s;
        }
      }
}

// ---------------------------------------------------------------------------
// MoE FFN2: out += gate * (act @ wo[e]). Wave tile 64x32.
// ---------------------------------------------------------------------------
__global__ void __launch_bounds__(256, 1)
k_moe_ffn2(const _Float16* __restrict__ act,
           const _Float16* __restrict__ wot,
           const int* __restrict__ cnt,
           const int* __restrict__ list,
           const float* __restrict__ gate,
           float* __restrict__ out) {
  int e = blockIdx.z;
  int c = cnt[e];
  if ((int)blockIdx.y * 128 >= c) return;
  const int* lst = list + e * M_;
  const _Float16* Wt = wot + (size_t)e * D_ * FH_;
  int lane = threadIdx.x & 31, wv = threadIdx.x >> 5;
  int m0 = blockIdx.y * 128 + (wv & 1) * 64;
  int n0 = blockIdx.x * 128 + (wv >> 1) * 32;
  v8f acc[4][2] = {};
  for (int k0 = 0; k0 < FH_; k0 += 32) {
    __builtin_prefetch(Wt + (size_t)(n0 + (lane & 15)) * FH_ + k0 + 64, 0, 1);
    v16h af[4], bf[2];
#pragma unroll
    for (int i = 0; i < 4; ++i)
      af[i] = frag_a_gather(act, FH_, lst, c, m0 + 16 * i, k0, lane);
#pragma unroll
    for (int j = 0; j < 2; ++j)
      bf[j] = frag_b(Wt, FH_, n0 + 16 * j, k0, lane);
#pragma unroll
    for (int i = 0; i < 4; ++i)
#pragma unroll
      for (int j = 0; j < 2; ++j) acc[i][j] = wmma_f16(af[i], bf[j], acc[i][j]);
  }
  int hb = lane >> 4, ln = lane & 15;
#pragma unroll
  for (int i = 0; i < 4; ++i)
#pragma unroll
    for (int j = 0; j < 2; ++j)
#pragma unroll
      for (int r = 0; r < 8; ++r) {
        int slot = m0 + i * 16 + r + 8 * hb;
        if (slot < c) {
          int tok = lst[slot];
          int n = n0 + j * 16 + ln;
          out[(size_t)tok * D_ + n] += gate[tok] * acc[i][j][r];
        }
      }
}

// ---------------------------------------------------------------------------
// Host-side launch
// ---------------------------------------------------------------------------
extern "C" void kernel_launch(void* const* d_in, const int* in_sizes, int n_in,
                              void* d_out, int out_size, void* d_ws,
                              size_t ws_size, hipStream_t stream) {
  const float* x      = (const float*)d_in[0];
  const float* ln1_w  = (const float*)d_in[1];
  const float* ln1_b  = (const float*)d_in[2];
  const float* ln2_w  = (const float*)d_in[3];
  const float* ln2_b  = (const float*)d_in[4];
  const float* w_qkv  = (const float*)d_in[5];
  const float* b_qkv  = (const float*)d_in[6];
  const float* w_out  = (const float*)d_in[7];
  const float* b_out  = (const float*)d_in[8];
  const float* w_rout = (const float*)d_in[9];
  const float* wi     = (const float*)d_in[10];
  const float* wo     = (const float*)d_in[11];

  float* out_x      = (float*)d_out;                       // (B,S,D)
  float* out_logits = (float*)d_out + (size_t)M_ * D_;     // (B,S,E)

  char* base = (char*)d_ws;
  size_t off = 0;
  auto carve = [&](size_t bytes) {
    char* p = base + off;
    off = (off + bytes + 255) & ~(size_t)255;
    return p;
  };
  _Float16* wq16 = (_Float16*)carve((size_t)N_QKV * D_ * 2);
  _Float16* wo16 = (_Float16*)carve((size_t)D_ * D_ * 2);
  _Float16* wit  = (_Float16*)carve((size_t)E_ * F_ * D_ * 2);
  _Float16* wot  = (_Float16*)carve((size_t)E_ * D_ * FH_ * 2);
  _Float16* hA   = (_Float16*)carve((size_t)M_ * D_ * 2);
  _Float16* qh   = (_Float16*)carve((size_t)M_ * D_ * 2);
  _Float16* kh   = (_Float16*)carve((size_t)M_ * D_ * 2);
  _Float16* vh   = (_Float16*)carve((size_t)M_ * D_ * 2);
  _Float16* o16  = (_Float16*)carve((size_t)M_ * D_ * 2);
  _Float16* h2   = (_Float16*)carve((size_t)M_ * D_ * 2);
  _Float16* act  = (_Float16*)carve((size_t)M_ * FH_ * 2);
  float*    gate = (float*)carve((size_t)M_ * 4);
  int*      cnt  = (int*)carve((size_t)E_ * 4);
  int*      list = (int*)carve((size_t)E_ * M_ * 4);
  (void)ws_size; (void)in_sizes; (void)n_in; (void)out_size;

  // 0) reset expert counters
  k_zero_counts<<<1, 32, 0, stream>>>(cnt);

  // 1) weight conversion / transposition to f16 (K-contiguous B layouts)
  k_cvt<<<(N_QKV * D_) / 256, 256, 0, stream>>>(w_qkv, wq16, N_QKV * D_);
  k_cvt<<<(D_ * D_) / 256, 256, 0, stream>>>(w_out, wo16, D_ * D_);
  k_cvt_T<<<dim3((D_ * F_) / 256, 1, E_), 256, 0, stream>>>(wi, wit, D_, F_);
  k_cvt_T<<<dim3((FH_ * D_) / 256, 1, E_), 256, 0, stream>>>(wo, wot, FH_, D_);

  // 2) LN1 + RoPE
  k_ln1_rope<<<M_, 256, 0, stream>>>(x, ln1_w, ln1_b, hA);

  // 3) QKV projection (64x32 wave tiles; scatter to head-major Q,K, V^T)
  k_gemm_qkv<<<dim3(N_QKV / 128, M_ / 128), 256, 0, stream>>>(hA, wq16, b_qkv,
                                                              qh, kh, vh);

  // 4) attention (2048 waves, 4 per block)
  k_attn<<<(B_ * H_ * (S_ / 16)) / 4, 128, 0, stream>>>(qh, kh, vh, o16);

  // 5) output projection + residual -> d_out (f32)
  k_gemm_out<<<dim3(D_ / 128, M_ / 128), 256, 0, stream>>>(o16, wo16, b_out, x,
                                                           out_x);

  // 6) LN2 + router (+gate, expert lists)
  k_ln2_router<<<M_, 256, 0, stream>>>(out_x, ln2_w, ln2_b, w_rout, h2,
                                       out_logits, gate, cnt, list);

  // 7) MoE FFN1 with fused SiLU-gate (32x(32+32) wave tiles)
  k_moe_ffn1<<<dim3(FH_ / 128, M_ / 64, E_), 256, 0, stream>>>(h2, wit, cnt,
                                                               list, act);

  // 8) MoE FFN2, gate-scaled accumulate into d_out (64x32 wave tiles)
  k_moe_ffn2<<<dim3(D_ / 128, M_ / 128, E_), 256, 0, stream>>>(act, wot, cnt,
                                                               list, gate,
                                                               out_x);
}
